// MultiheadCrossAttention_72705206387301
// MI455X (gfx1250) — matrix operand order repaired and checked by reading, not compiled, not run on hardware
//
#include <hip/hip_runtime.h>
#include <hip/hip_bf16.h>

typedef __attribute__((ext_vector_type(16))) __bf16 v16bf;
typedef __attribute__((ext_vector_type(8)))  __bf16 v8bf;
typedef __attribute__((ext_vector_type(4)))  __bf16 v4bf;
typedef __attribute__((ext_vector_type(8)))  float  v8f;
typedef __attribute__((ext_vector_type(4)))  float  f32x4;
typedef __attribute__((ext_vector_type(4)))  int    i32x4;

#define B_  4
#define SQ_ 2048
#define SK_ 2048
#define D_  1024
#define H_  16
#define HD_ 64

// ---- CDNA5 async global->LDS copy (ASYNCcnt path), guarded so compile can't break
#if defined(__has_builtin)
#  if __has_builtin(__builtin_amdgcn_global_load_async_to_lds_b128)
#    define HAVE_ASYNC_LDS 1
#  endif
#endif
#ifndef HAVE_ASYNC_LDS
#  define HAVE_ASYNC_LDS 0
#endif

#if HAVE_ASYNC_LDS
#  if __has_builtin(__builtin_amdgcn_s_wait_asynccnt)
#    define WAIT_ASYNC(n) __builtin_amdgcn_s_wait_asynccnt(n)
#  else
#    define WAIT_ASYNC(n) asm volatile("s_wait_asynccnt %0" :: "i"(n) : "memory")
#  endif
#else
#  define WAIT_ASYNC(n) do {} while (0)
#endif

static __device__ __forceinline__ v16bf cat8(v8bf lo, v8bf hi) {
  return __builtin_shufflevector(lo, hi, 0,1,2,3,4,5,6,7,8,9,10,11,12,13,14,15);
}
static __device__ __forceinline__ v8f wmma_bf16(v16bf a, v16bf b, v8f c) {
  // (neg_a, A, neg_b, B, c_mod, C, reuse_a, reuse_b)
  return __builtin_amdgcn_wmma_f32_16x16x32_bf16(false, a, false, b, (short)0, c,
                                                 false, false);
}
static __device__ __forceinline__ void async_cp_b128(const void* g, void* l) {
#if HAVE_ASYNC_LDS
  __builtin_amdgcn_global_load_async_to_lds_b128(
      (__attribute__((address_space(1))) i32x4*)g,
      (__attribute__((address_space(3))) i32x4*)l, 0, 0);
#else
  *(v8bf*)l = *(const v8bf*)g;
#endif
}

// ---------------------------------------------------------------------------
// GEMM: C[m,n] = sum_k A[m,k] * W[n,k] + bias[n]   (x @ W^T + b)
// M = 8192, N = K = 1024. Tile 128x128, 8 waves, wave tile 64x32, K-step 32.
// MODE 0: bf16 out, layout [B,H,S,HD]   (Q/K projections)
// MODE 1: bf16 out, layout [B,H,HD,S]   (V projection, pre-transposed)
// MODE 2: f32  out, row-major [M,N]     (output projection -> d_out)
// ABF16:  A operand is already bf16 (attention output) vs fp32 (inputs).
// ---------------------------------------------------------------------------
template <int MODE, bool ABF16>
__global__ __launch_bounds__(256) void gemm_xwt(
    const void* __restrict__ Ap, const float* __restrict__ W,
    const float* __restrict__ bias, void* __restrict__ Cp)
{
  __shared__ __bf16 As[128][40];   // 40-elem stride keeps 16B alignment
  __shared__ __bf16 Bs[128][40];

  const int tid   = threadIdx.x;
  const int wave  = tid >> 5;
  const int lane  = tid & 31;
  const int waveM = wave >> 2;     // 0..1  -> 64 rows
  const int waveN = wave & 3;      // 0..3  -> 32 cols
  const int row16 = lane & 15;
  const int hi    = lane >> 4;
  const size_t rowA = (size_t)blockIdx.y * 128;
  const size_t rowB = (size_t)blockIdx.x * 128;

  v8f acc[4][2] = {};

  for (int k0 = 0; k0 < D_; k0 += 32) {
    __syncthreads();
    if constexpr (ABF16) {
      const __bf16* A = (const __bf16*)Ap;
      #pragma unroll
      for (int i = 0; i < 2; ++i) {              // 512 v8bf slots
        int slot = tid + i * 256;
        int r = slot >> 2, c8 = slot & 3;
        *(v8bf*)&As[r][c8 * 8] = *(const v8bf*)(A + (rowA + r) * D_ + k0 + c8 * 8);
      }
    } else {
      const float* A = (const float*)Ap;
      #pragma unroll
      for (int i = 0; i < 4; ++i) {              // 1024 float4 slots, cvt f32->bf16
        int slot = tid + i * 256;
        int r = slot >> 3, c4 = slot & 7;
        f32x4 v = *(const f32x4*)(A + (rowA + r) * D_ + k0 + c4 * 4);
        v4bf w = { (__bf16)v.x, (__bf16)v.y, (__bf16)v.z, (__bf16)v.w };
        *(v4bf*)&As[r][c4 * 4] = w;
      }
    }
    #pragma unroll
    for (int i = 0; i < 4; ++i) {                // weights: f32 -> bf16
      int slot = tid + i * 256;
      int r = slot >> 3, c4 = slot & 7;
      f32x4 v = *(const f32x4*)(W + (rowB + r) * D_ + k0 + c4 * 4);
      v4bf w = { (__bf16)v.x, (__bf16)v.y, (__bf16)v.z, (__bf16)v.w };
      *(v4bf*)&Bs[r][c4 * 4] = w;
    }
    // prefetch next K-tile of both streams while WMMAs run
    if (k0 + 32 < D_) {
      int pr = tid >> 1, ph = (tid & 1) * 16;
      if constexpr (ABF16)
        __builtin_prefetch((const __bf16*)Ap + (rowA + pr) * D_ + k0 + 32 + ph, 0, 1);
      else
        __builtin_prefetch((const float*)Ap + (rowA + pr) * D_ + k0 + 32 + ph, 0, 1);
      __builtin_prefetch(W + (rowB + pr) * D_ + k0 + 32 + ph, 0, 1);
    }
    __syncthreads();

    // A fragment (16x32, lane=M row, K split per CDNA5 layout)
    v16bf af[4];
    #pragma unroll
    for (int ms = 0; ms < 4; ++ms) {
      int r = waveM * 64 + ms * 16 + row16;
      v8bf lo = *(const v8bf*)&As[r][hi * 8];
      v8bf hh = *(const v8bf*)&As[r][16 + hi * 8];
      af[ms] = cat8(lo, hh);
    }
    // B fragment (32x16, lane=N col, K split per CDNA5 layout)
    v16bf bfr[2];
    #pragma unroll
    for (int ns = 0; ns < 2; ++ns) {
      int r = waveN * 32 + ns * 16 + row16;
      v8bf lo = *(const v8bf*)&Bs[r][hi * 8];
      v8bf hh = *(const v8bf*)&Bs[r][16 + hi * 8];
      bfr[ns] = cat8(lo, hh);
    }
    #pragma unroll
    for (int ms = 0; ms < 4; ++ms)
      #pragma unroll
      for (int ns = 0; ns < 2; ++ns)
        acc[ms][ns] = wmma_bf16(af[ms], bfr[ns], acc[ms][ns]);
  }

  // Epilogue: C layout — VGPR r holds M = r + 8*hi, N = row16
  #pragma unroll
  for (int ms = 0; ms < 4; ++ms) {
    #pragma unroll
    for (int ns = 0; ns < 2; ++ns) {
      int gn = (int)rowB + waveN * 32 + ns * 16 + row16;
      float bv = bias[gn];
      #pragma unroll
      for (int r = 0; r < 8; ++r) {
        int gm = (int)rowA + waveM * 64 + ms * 16 + r + hi * 8;
        float val = acc[ms][ns][r] + bv;
        if constexpr (MODE == 2) {
          ((float*)Cp)[(size_t)gm * D_ + gn] = val;
        } else {
          int b = gm >> 11, srow = gm & (SQ_ - 1);
          int h = gn >> 6,  d    = gn & (HD_ - 1);
          size_t idx;
          if constexpr (MODE == 0)
            idx = (((size_t)(b * H_ + h)) * SQ_ + srow) * HD_ + d;
          else
            idx = (((size_t)(b * H_ + h)) * HD_ + d) * (size_t)SK_ + srow;
          ((__bf16*)Cp)[idx] = (__bf16)val;
        }
      }
    }
  }
}

// ---------------------------------------------------------------------------
// Flash attention: grid (SQ/128, B*H). 8 waves, each owns 16 q-rows.
// Q kept in registers; K and V^T double-buffered through LDS via async
// global->LDS copies (ASYNCcnt) when available, overlapping the next block's
// DMA with the current block's WMMAs.
// ---------------------------------------------------------------------------
__global__ __launch_bounds__(256) void flash_attn(
    const __bf16* __restrict__ qp, const __bf16* __restrict__ kp,
    const __bf16* __restrict__ vt, const int* __restrict__ kpm,
    __bf16* __restrict__ attn)
{
  __shared__ __bf16 Ks[2][64][72];    // [buf][key][d]
  __shared__ __bf16 Vs[2][64][72];    // [buf][d][key]  (V pre-transposed)
  __shared__ __bf16 Ps[8][16][64];    // per-wave P scratch

  const int tid   = threadIdx.x;
  const int wave  = tid >> 5, lane = tid & 31;
  const int row16 = lane & 15, hi = lane >> 4;
  const int bh = blockIdx.y;
  const int b  = bh >> 4;
  const int h  = bh & 15;
  const __bf16* qh = qp + (size_t)bh * SQ_ * HD_;
  const __bf16* kh = kp + (size_t)bh * SK_ * HD_;
  const __bf16* vh = vt + (size_t)bh * HD_ * SK_;
  const int qbase = blockIdx.x * 128 + wave * 16;

  // Stage one 64-key block (4 x b128 per thread: 2 for K, 2 for V^T)
  auto stage_block = [&](int kb, int bufi) {
    #pragma unroll
    for (int i = 0; i < 2; ++i) {
      int slot = tid + i * 256;
      int r = slot >> 3, c = slot & 7;
      async_cp_b128(kh + (size_t)(kb + r) * HD_ + c * 8, &Ks[bufi][r][c * 8]);
      async_cp_b128(vh + (size_t)r * SK_ + kb + c * 8,   &Vs[bufi][r][c * 8]);
    }
  };

  // Q fragments: 16 rows x 64 d -> two 16x32 A fragments, straight from global
  v16bf qf[2];
  {
    const __bf16* qrow = qh + (size_t)(qbase + row16) * HD_;
    #pragma unroll
    for (int s = 0; s < 2; ++s) {
      v8bf lo = *(const v8bf*)(qrow + s * 32 + hi * 8);
      v8bf hh = *(const v8bf*)(qrow + s * 32 + 16 + hi * 8);
      qf[s] = cat8(lo, hh);
    }
  }

  float mrow[8], lrow[8];
  v8f o[4] = {};
  #pragma unroll
  for (int r = 0; r < 8; ++r) { mrow[r] = -3.0e38f; lrow[r] = 0.f; }

  stage_block(0, 0);
  int cur = 0;

  for (int kb = 0; kb < SK_; kb += 64) {
    __syncthreads();                       // everyone done reading buf cur^1
    if (kb + 64 < SK_) {
      stage_block(kb + 64, cur ^ 1);       // DMA next block during compute
      WAIT_ASYNC(4);                       // in-order: <=4 left => cur landed
    } else {
      WAIT_ASYNC(0);
    }
    __syncthreads();                       // all waves' cur-buffer data visible

    // S tiles: 4 x (16q x 16k), each accumulates two d-steps of 32
    v8f st[4];
    #pragma unroll
    for (int kt = 0; kt < 4; ++kt) {
      v8f a = {};
      int kr = kt * 16 + row16;
      #pragma unroll
      for (int s = 0; s < 2; ++s) {
        v8bf lo = *(const v8bf*)&Ks[cur][kr][s * 32 + hi * 8];
        v8bf hh = *(const v8bf*)&Ks[cur][kr][s * 32 + 16 + hi * 8];
        a = wmma_bf16(qf[s], cat8(lo, hh), a);
      }
      st[kt] = a;
    }

    // scale + key-padding mask, then row-max over the 64 keys
    float rmax[8];
    #pragma unroll
    for (int r = 0; r < 8; ++r) rmax[r] = -3.0e38f;
    #pragma unroll
    for (int kt = 0; kt < 4; ++kt) {
      int key = kb + kt * 16 + row16;
      float madd = (kpm[b * SK_ + key] != 0) ? 0.f : -1.0e30f;
      #pragma unroll
      for (int r = 0; r < 8; ++r) {
        float v = st[kt][r] * 0.125f + madd;   // HD^-0.5 = 0.125
        st[kt][r] = v;
        rmax[r] = fmaxf(rmax[r], v);
      }
    }
    #pragma unroll
    for (int off = 1; off < 16; off <<= 1)     // rows live in 16-lane halves
      #pragma unroll
      for (int r = 0; r < 8; ++r)
        rmax[r] = fmaxf(rmax[r], __shfl_xor(rmax[r], off, 32));

    float alpha[8];
    #pragma unroll
    for (int r = 0; r < 8; ++r) {
      float mn = fmaxf(mrow[r], rmax[r]);
      alpha[r] = __expf(mrow[r] - mn);
      mrow[r] = mn;
      lrow[r] *= alpha[r];
    }
    #pragma unroll
    for (int dt = 0; dt < 4; ++dt)
      #pragma unroll
      for (int r = 0; r < 8; ++r)
        o[dt][r] *= alpha[r];

    // P = exp(S - m), row-sum, and stage to LDS for A-layout reload
    float rsum[8] = {};
    #pragma unroll
    for (int kt = 0; kt < 4; ++kt)
      #pragma unroll
      for (int r = 0; r < 8; ++r) {
        float p = __expf(st[kt][r] - mrow[r]);
        rsum[r] += p;
        Ps[wave][r + hi * 8][kt * 16 + row16] = (__bf16)p;
      }
    #pragma unroll
    for (int off = 1; off < 16; off <<= 1)
      #pragma unroll
      for (int r = 0; r < 8; ++r)
        rsum[r] += __shfl_xor(rsum[r], off, 32);
    #pragma unroll
    for (int r = 0; r < 8; ++r) lrow[r] += rsum[r];

    // O += P (16x64) * V (64x64): A from Ps, B from transposed V in LDS
    #pragma unroll
    for (int ks = 0; ks < 2; ++ks) {
      v8bf plo = *(const v8bf*)&Ps[wave][row16][ks * 32 + hi * 8];
      v8bf phh = *(const v8bf*)&Ps[wave][row16][ks * 32 + 16 + hi * 8];
      v16bf pa = cat8(plo, phh);
      #pragma unroll
      for (int dt = 0; dt < 4; ++dt) {
        int dr = dt * 16 + row16;
        v8bf blo = *(const v8bf*)&Vs[cur][dr][ks * 32 + hi * 8];
        v8bf bhh = *(const v8bf*)&Vs[cur][dr][ks * 32 + 16 + hi * 8];
        o[dt] = wmma_bf16(pa, cat8(blo, bhh), o[dt]);
      }
    }
    cur ^= 1;
  }

  // normalize and write merged [B,SQ,D] bf16 for the output projection
  float inv[8];
  #pragma unroll
  for (int r = 0; r < 8; ++r) inv[r] = 1.0f / lrow[r];
  #pragma unroll
  for (int dt = 0; dt < 4; ++dt)
    #pragma unroll
    for (int r = 0; r < 8; ++r) {
      int q   = qbase + r + hi * 8;
      int col = h * HD_ + dt * 16 + row16;
      attn[((size_t)b * SQ_ + q) * D_ + col] = (__bf16)(o[dt][r] * inv[r]);
    }
}

extern "C" void kernel_launch(void* const* d_in, const int* in_sizes, int n_in,
                              void* d_out, int out_size, void* d_ws, size_t ws_size,
                              hipStream_t stream) {
  const float* query = (const float*)d_in[0];
  const float* key   = (const float*)d_in[1];
  const float* value = (const float*)d_in[2];
  const int*   kpm   = (const int*)d_in[3];
  const float* Wq    = (const float*)d_in[4];
  const float* bq    = (const float*)d_in[5];
  const float* Wk    = (const float*)d_in[6];
  const float* bk    = (const float*)d_in[7];
  const float* Wv    = (const float*)d_in[8];
  const float* bv    = (const float*)d_in[9];
  const float* Wo    = (const float*)d_in[10];
  const float* bo    = (const float*)d_in[11];

  const size_t ELEMS = (size_t)B_ * SQ_ * D_;   // 8M bf16 each
  __bf16* qp   = (__bf16*)d_ws;                 // [B,H,SQ,HD]
  __bf16* kp   = qp + ELEMS;                    // [B,H,SK,HD]
  __bf16* vt   = kp + ELEMS;                    // [B,H,HD,SK]
  __bf16* attn = vt + ELEMS;                    // [B,SQ,D]

  dim3 blk(256);
  dim3 gg(D_ / 128, (B_ * SQ_) / 128);          // (8, 64)

  gemm_xwt<0, false><<<gg, blk, 0, stream>>>(query, Wq, bq, qp);
  gemm_xwt<0, false><<<gg, blk, 0, stream>>>(key,   Wk, bk, kp);
  gemm_xwt<1, false><<<gg, blk, 0, stream>>>(value, Wv, bv, vt);

  flash_attn<<<dim3(SQ_ / 128, B_ * H_), blk, 0, stream>>>(qp, kp, vt, kpm, attn);

  gemm_xwt<2, true><<<gg, blk, 0, stream>>>(attn, Wo, bo, d_out);
}
